// VectorQuantizerEMA_10900626997675
// MI455X (gfx1250) — matrix operand order repaired and checked by reading, not compiled, not run on hardware
//
#include <hip/hip_runtime.h>

typedef _Float16 v16h __attribute__((ext_vector_type(16)));
typedef float     v8f __attribute__((ext_vector_type(8)));
typedef unsigned int u32x4 __attribute__((ext_vector_type(4)));
typedef int          i32x8 __attribute__((ext_vector_type(8)));
typedef int          i32x4 __attribute__((ext_vector_type(4)));

#define K_CB      1024
#define D_CB      64
#define T_LEN     4096
#define B_SZ      32
#define N_TOT     (B_SZ * T_LEN)          /* 131072 */
#define ZQ_ELEMS  (B_SZ * D_CB * T_LEN)   /* 8388608 */
#define DECAYF    0.99f
#define ONE_M_DEC 0.01f
#define EPSV      1e-5f
#define LDS_STRIDE 72        /* halves: 128B data + 16B pad per row -> bank-conflict spread */
#define ELDS_BYTES (K_CB * LDS_STRIDE * 2)   /* 147456 dynamic LDS for padded codebook */

// ---------------------------------------------------------------------------
// K0: convert embedding to f16, compute ||e||^2, zero accumulators + loss slot
// grid: K_CB blocks x D_CB threads (block = codebook row)
// ---------------------------------------------------------------------------
__global__ void vq_prep(const float* __restrict__ emb,
                        _Float16* __restrict__ eHalf,
                        float* __restrict__ eNorm,
                        float* __restrict__ counts,
                        float* __restrict__ embSum,
                        float* __restrict__ lossSlot) {
    const int k = blockIdx.x;
    const int d = threadIdx.x;
    const int o = k * D_CB + d;
    const float v = emb[o];
    eHalf[o]  = (_Float16)v;
    embSum[o] = 0.0f;

    __shared__ float red[D_CB];
    red[d] = v * v;
    __syncthreads();
    for (int s = D_CB >> 1; s > 0; s >>= 1) {
        if (d < s) red[d] += red[d + s];
        __syncthreads();
    }
    if (d == 0) {
        eNorm[k]  = red[0];
        counts[k] = 0.0f;
        if (k == 0) lossSlot[0] = 0.0f;
    }
}

// ---------------------------------------------------------------------------
// K1: distance GEMM via v_wmma_f32_16x16x32_f16 + per-row argmin.
// The full f16 codebook (128KB) is DMA'd into LDS by the Tensor Data Mover
// with hardware row-padding (32 DW data + 4 DW pad = 72-half stride), so the
// hot loop reads B fragments from LDS instead of hammering L2 8x per block.
// grid: N_TOT/128 blocks x 256 threads (8 waves x 16 rows), 147456B dyn LDS
// ---------------------------------------------------------------------------
__global__ void __launch_bounds__(256)
vq_argmin(const float* __restrict__ z,
          const _Float16* __restrict__ eHalf,
          const float* __restrict__ eNorm,
          int* __restrict__ idxI,
          float* __restrict__ idxF) {
    extern __shared__ _Float16 eLds[];             // K_CB rows x LDS_STRIDE halves
    __shared__ _Float16 zLds[128 * LDS_STRIDE];
    __shared__ float    sNorm[K_CB];

    const int tid = threadIdx.x;

#if __has_builtin(__builtin_amdgcn_tensor_load_to_lds)
    // --- TDM: async copy eHalf[1024][64] (f16) -> padded eLds ---------------
    if (tid == 0) {
        const unsigned long long ga = (unsigned long long)(uintptr_t)eHalf;
        const unsigned ldsOff = (unsigned)(uintptr_t)eLds;   // low 32 bits = LDS offset
        u32x4 g0;
        g0[0] = 1u;                                   // count=1 (valid user descriptor)
        g0[1] = ldsOff;                               // lds_addr
        g0[2] = (unsigned)(ga & 0xFFFFFFFFu);         // global_addr[31:0]
        g0[3] = (unsigned)((ga >> 32) & 0x01FFFFFFu)  // global_addr[56:32]
                | 0x80000000u;                        // type=2 ("image")
        i32x8 g1;
        g1[0] = (int)0x07110000u;  // data_size=2B | pad_enable | pad_interval=32DW | pad_amount=4DW
        g1[1] = (int)(64u   << 16); // tensor_dim0 = 64   (bits 79:48, low half)
        g1[2] = (int)(1024u << 16); // tensor_dim1 = 1024 (bits 111:80, low half)
        g1[3] = (int)(64u   << 16); // tile_dim0 = 64     (bits 127:112)
        g1[4] = (int)1024;          // tile_dim1 = 1024; tile_dim2 = 0
        g1[5] = (int)64;            // tensor_dim0_stride = 64 (bits 207:160, low 32)
        g1[6] = 0;
        g1[7] = 0;
        const i32x4 gz4 = {0, 0, 0, 0};
        const i32x8 gz8 = {0, 0, 0, 0, 0, 0, 0, 0};
        // 6-arg form (clang-23 / therock toolchain probe-confirmed arity)
        __builtin_amdgcn_tensor_load_to_lds(g0, g1, gz4, gz4, gz8, 0);
    }
#else
    for (int i = tid; i < K_CB * D_CB; i += 256) {
        const int r = i >> 6, c = i & 63;
        eLds[r * LDS_STRIDE + c] = eHalf[i];
    }
#endif

    const int n0  = blockIdx.x * 128;        // 128 consecutive flat rows, one batch b
    const int b   = n0 >> 12;                // T = 4096
    const int t0  = n0 & (T_LEN - 1);
    const float* zb = z + (size_t)b * D_CB * T_LEN;

    // Stage 128 rows x 64 dims of z into LDS as f16 (coalesced along t);
    // overlaps with the TDM transfer.
    #pragma unroll
    for (int it = 0; it < 32; ++it) {
        const int e = it * 256 + tid;        // 8192 elements
        const int d = e >> 7;
        const int r = e & 127;
        zLds[r * LDS_STRIDE + d] = (_Float16)zb[d * T_LEN + t0 + r];
    }
    for (int i = tid; i < K_CB; i += 256) sNorm[i] = eNorm[i];

#if __has_builtin(__builtin_amdgcn_tensor_load_to_lds)
    if (tid == 0) __builtin_amdgcn_s_wait_tensorcnt(0);
#endif
    __syncthreads();

    const int l    = tid & 31;               // lane in wave32
    const int wv   = tid >> 5;               // wave id: 8 waves, 16 rows each
    const int half = l >> 4;                 // 0: lanes 0-15, 1: lanes 16-31
    const int m    = l & 15;                 // A-matrix row M = lane%16

    // A fragments (contiguous K order, matching B):
    //  lanes 0-15 : halves 0-15 -> K=kc..kc+15
    //  lanes 16-31: shifted by +16
    const _Float16* zr = &zLds[(wv * 16 + m) * LDS_STRIDE];
    const int kb = half * 16;
    v16h a0, a1;
    #pragma unroll
    for (int h = 0; h < 8; ++h) {
        a0[h]     = zr[kb + h];
        a0[h + 8] = zr[kb + 8 + h];
        a1[h]     = zr[kb + 32 + h];
        a1[h + 8] = zr[kb + 40 + h];
    }

    float minVal[8];
    int   minIdx[8];
    #pragma unroll
    for (int j = 0; j < 8; ++j) { minVal[j] = 3.402823466e+38f; minIdx[j] = 0; }

    const int dB = half * 16;                // B layout: lanes 0-15 hold d 0-15, lanes 16-31 hold 16-31
    #pragma unroll 2
    for (int tile = 0; tile < K_CB / 16; ++tile) {
        const int col = tile * 16 + m;       // codebook row for this lane (C: N = lane%16)
        const _Float16* ep = &eLds[col * LDS_STRIDE + dB];
        v16h b0, b1;
        #pragma unroll
        for (int h = 0; h < 16; ++h) { b0[h] = ep[h]; b1[h] = ep[32 + h]; }

        v8f c = {};
        c = __builtin_amdgcn_wmma_f32_16x16x32_f16(false, a0, false, b0, (short)0, c, false, false);
        c = __builtin_amdgcn_wmma_f32_16x16x32_f16(false, a1, false, b1, (short)0, c, false, false);

        const float en = sNorm[col];
        #pragma unroll
        for (int j = 0; j < 8; ++j) {
            const float dist = en - 2.0f * c[j];   // ||e||^2 - 2 z.e (||z||^2 constant per row)
            if (dist < minVal[j]) { minVal[j] = dist; minIdx[j] = col; }
        }
    }

    // Cross-lane argmin over the 16 lanes sharing each row M (first-index tie-break)
    #pragma unroll
    for (int msk = 1; msk <= 8; msk <<= 1) {
        #pragma unroll
        for (int j = 0; j < 8; ++j) {
            const float ov = __shfl_xor(minVal[j], msk, 32);
            const int   oi = __shfl_xor(minIdx[j], msk, 32);
            if (ov < minVal[j] || (ov == minVal[j] && oi < minIdx[j])) {
                minVal[j] = ov; minIdx[j] = oi;
            }
        }
    }

    if (m == 0) {   // lanes 0 and 16 hold rows j and j+8 respectively
        #pragma unroll
        for (int j = 0; j < 8; ++j) {
            const int row = j + half * 8;    // C layout: VGPR j -> M=j (lanes<16) / M=j+8
            const int n   = n0 + wv * 16 + row;
            idxI[n] = minIdx[j];
            idxF[n] = (float)minIdx[j];
        }
    }
}

// ---------------------------------------------------------------------------
// K2: segment sums (counts + embed_sum) via float atomics; z re-read hits L2
// grid: N_TOT/256 x 256
// ---------------------------------------------------------------------------
__global__ void vq_scatter(const float* __restrict__ z,
                           const int* __restrict__ idxI,
                           float* __restrict__ counts,
                           float* __restrict__ embSum) {
    const int n = blockIdx.x * 256 + threadIdx.x;
    const int idx = idxI[n];
    atomicAdd(&counts[idx], 1.0f);
    const int b = n >> 12;
    const int t = n & (T_LEN - 1);
    const float* zb = z + (size_t)b * D_CB * T_LEN + t;
    float* dst = embSum + idx * D_CB;
    #pragma unroll
    for (int d = 0; d < D_CB; ++d)
        atomicAdd(&dst[d], zb[d * T_LEN]);
}

// ---------------------------------------------------------------------------
// K3: EMA update + Laplace smoothing + new embedding. 1 block x 1024 threads.
// ---------------------------------------------------------------------------
__global__ void __launch_bounds__(1024)
vq_ema(const float* __restrict__ clusterSize,
       const float* __restrict__ embedAvg,
       const float* __restrict__ counts,
       const float* __restrict__ embSum,
       float* __restrict__ newEmb) {
    const int k = threadIdx.x;
    const float ncs = clusterSize[k] * DECAYF + ONE_M_DEC * counts[k];

    __shared__ float red[1024];
    red[k] = ncs;
    __syncthreads();
    for (int s = 512; s > 0; s >>= 1) {
        if (k < s) red[k] += red[k + s];
        __syncthreads();
    }
    const float nsum = red[0];
    const float smoothed = (ncs + EPSV) / (nsum + (float)K_CB * EPSV) * nsum;
    const float inv = 1.0f / smoothed;
    #pragma unroll
    for (int d = 0; d < D_CB; ++d) {
        const int o = k * D_CB + d;
        newEmb[o] = (embedAvg[o] * DECAYF + ONE_M_DEC * embSum[o]) * inv;
    }
}

// ---------------------------------------------------------------------------
// K4: z_q gather (straight-through => z_q == new_embedding[idx]) + loss
// grid: N_TOT/128 x 256
// ---------------------------------------------------------------------------
__global__ void __launch_bounds__(256)
vq_output(const float* __restrict__ z,
          const int* __restrict__ idxI,
          const float* __restrict__ newEmb,
          float* __restrict__ zq,
          float* __restrict__ lossSlot) {
    const int tid = threadIdx.x;
    const int n0  = blockIdx.x * 128;
    const int b   = n0 >> 12;
    const int t0  = n0 & (T_LEN - 1);

    __shared__ int sIdx[128];
    if (tid < 128) sIdx[tid] = idxI[n0 + tid];
    __syncthreads();

    const size_t base = (size_t)b * D_CB * T_LEN + t0;
    float acc = 0.0f;
    #pragma unroll
    for (int it = 0; it < 32; ++it) {
        const int e = it * 256 + tid;
        const int d = e >> 7;
        const int r = e & 127;
        const float zv = z[base + d * T_LEN + r];
        const float q  = newEmb[sIdx[r] * D_CB + d];
        zq[base + d * T_LEN + r] = q;
        const float diff = zv - q;
        acc += diff * diff;
    }

    __shared__ float red[256];
    red[tid] = acc;
    __syncthreads();
    for (int s = 128; s > 0; s >>= 1) {
        if (tid < s) red[tid] += red[tid + s];
        __syncthreads();
    }
    if (tid == 0)
        atomicAdd(lossSlot, red[0] * (0.25f / ((float)N_TOT * (float)D_CB)));
}

// ---------------------------------------------------------------------------
extern "C" void kernel_launch(void* const* d_in, const int* in_sizes, int n_in,
                              void* d_out, int out_size, void* d_ws, size_t ws_size,
                              hipStream_t stream) {
    const float* z           = (const float*)d_in[0];
    const float* embedding   = (const float*)d_in[1];
    const float* clusterSize = (const float*)d_in[2];
    const float* embedAvg    = (const float*)d_in[3];

    float* out = (float*)d_out;
    char*  ws  = (char*)d_ws;

    // workspace layout (bytes)
    _Float16* eHalf  = (_Float16*)(ws);                 // 1024*64*2 = 131072
    float*    eNorm  = (float*)(ws + 131072);           // 4096
    float*    counts = (float*)(ws + 135168);           // 4096
    float*    embSum = (float*)(ws + 139264);           // 262144
    float*    newEmb = (float*)(ws + 401408);           // 262144
    int*      idxI   = (int*)  (ws + 663552);           // 524288  (total ~1.16 MB)

    float* zq       = out;
    float* lossSlot = out + ZQ_ELEMS;
    float* idxF     = out + ZQ_ELEMS + 1;

    vq_prep   <<<K_CB,        D_CB, 0,          stream>>>(embedding, eHalf, eNorm, counts, embSum, lossSlot);
    vq_argmin <<<N_TOT / 128, 256,  ELDS_BYTES, stream>>>(z, eHalf, eNorm, idxI, idxF);
    vq_scatter<<<N_TOT / 256, 256,  0,          stream>>>(z, idxI, counts, embSum);
    vq_ema    <<<1,           1024, 0,          stream>>>(clusterSize, embedAvg, counts, embSum, newEmb);
    vq_output <<<N_TOT / 128, 256,  0,          stream>>>(z, idxI, newEmb, zq, lossSlot);
}
